// GLN_56006373539841
// MI455X (gfx1250) — matrix operand order repaired and checked by reading, not compiled, not run on hardware
//
#include <hip/hip_runtime.h>
#include <stdint.h>

// ---------------------------------------------------------------------------
// GLN on MI455X (gfx1250): HBM-bound (~400MB unique traffic vs ~680 GFLOP),
// so: dense bf16 WMMA GEMM with the 1-of-8 context selection fused into the
// epilogue (never materialize y[B,O,NC] = 268MB), fp32->bf16 conversion done
// on the global->LDS fill path, f32 accumulation in v_wmma_f32_16x16x32_bf16.
// ---------------------------------------------------------------------------

typedef __attribute__((ext_vector_type(16))) __bf16 v16bf;
typedef __attribute__((ext_vector_type(8)))  float  v8f;

#define BM 128           // block tile M (batch rows)
#define BN 128           // block tile N (weight rows = O*NC or O*3 columns)
#define BK 32            // k-slice per WMMA step (bf16 K=32)
#define LDK 40           // BK + 8 bf16 pad: row stride 80B, keeps 16B alignment,
                         // spreads ds_load_b128 across banks

// A-matrix 16x32 bf16 fragment (ISA 7.12.2): lanes 0-15 hold row M=l16 with
// K[0..7] in V0-3 and K[16..23] in V4-7; lanes 16-31 hold K[8..15], K[24..31].
// => two 16B chunks per lane at k = half*8 and half*8 + 16.
__device__ __forceinline__ v16bf ld_frag_A(const __bf16* row, int half) {
  v16bf a;
  ((uint4*)&a)[0] = *(const uint4*)(row + half * 8);
  ((uint4*)&a)[1] = *(const uint4*)(row + half * 8 + 16);
  return a;
}

// B-matrix 32x16 bf16 (column N=l16 per lane), same k order as A:
// lanes 0-15 hold K[0..7], K[16..23]; lanes 16-31 hold K[8..15], K[24..31].
// => two 16B chunks per lane at k = half*8 and half*8 + 16.
// Since we store the W tile as [n][k] row-major in LDS (W is already [rows,K]
// row-major in memory), this computes x @ W^T with no transposes anywhere.
__device__ __forceinline__ v16bf ld_frag_B(const __bf16* row, int half) {
  v16bf b;
  ((uint4*)&b)[0] = *(const uint4*)(row + half * 8);
  ((uint4*)&b)[1] = *(const uint4*)(row + half * 8 + 16);
  return b;
}

// MODE 0: GLN layer. out[b][o] = acc[b][n] + bias[n]  iff (n&7)==ctx[b][o], o=n>>3.
// MODE 1: context bits. bits[b][n] = acc[b][n] > T[n]  (n = o*3 + s).
template <int MODE>
__global__ __launch_bounds__(256)
void gln_gemm(const float* __restrict__ Ag,   // [B, K] activations (fp32)
              const float* __restrict__ Wg,   // [N, K] weights/hyperplanes (fp32)
              const float* __restrict__ bias, // [N] or nullptr       (MODE 0)
              const uint8_t* __restrict__ ctx,// [B, O] context index (MODE 0)
              const float* __restrict__ Tg,   // [N] thresholds       (MODE 1)
              uint8_t* __restrict__ bits,     // [B, N] bit output    (MODE 1)
              float* __restrict__ out,        // [B, O] selected out  (MODE 0)
              int K, int N, int O)
{
  __shared__ __bf16 As[2][BM * LDK];
  __shared__ __bf16 Bs[2][BN * LDK];

  const int tid  = threadIdx.x;
  const int lane = tid & 31;
  const int wave = tid >> 5;
  const int wm   = wave >> 2;   // 2 wave rows  -> 64 M each
  const int wn   = wave & 3;    // 4 wave cols  -> 32 N each
  const int half = lane >> 4;
  const int l16  = lane & 15;
  const int bx = blockIdx.x, by = blockIdx.y;

  v8f acc[4][2];
#pragma unroll
  for (int i = 0; i < 4; ++i)
#pragma unroll
    for (int j = 0; j < 2; ++j)
      acc[i][j] = (v8f){0.f, 0.f, 0.f, 0.f, 0.f, 0.f, 0.f, 0.f};

  const size_t rowA0 = (size_t)by * BM;
  const size_t rowB0 = (size_t)bx * BN;

  // 256 threads move a 128x32 fp32 tile per operand: 4 float4 each.
  float4 ra[4], rb[4];
  auto gload = [&](int kt) {
#pragma unroll
    for (int i = 0; i < 4; ++i) {
      const int f = tid + i * 256;
      const int r = f >> 3, c4 = f & 7;          // 8 float4 per 32-float row
      ra[i] = ((const float4*)(Ag + (rowA0 + r) * K + (size_t)kt * BK))[c4];
      rb[i] = ((const float4*)(Wg + (rowB0 + r) * K + (size_t)kt * BK))[c4];
    }
  };
  auto lstore = [&](int buf) {                    // fp32 -> bf16 on the fill path
#pragma unroll
    for (int i = 0; i < 4; ++i) {
      const int f = tid + i * 256;
      const int r = f >> 3, c4 = f & 7;
      union { __bf16 h[4]; uint2 u; } ca, cb;
      ca.h[0] = (__bf16)ra[i].x; ca.h[1] = (__bf16)ra[i].y;
      ca.h[2] = (__bf16)ra[i].z; ca.h[3] = (__bf16)ra[i].w;
      cb.h[0] = (__bf16)rb[i].x; cb.h[1] = (__bf16)rb[i].y;
      cb.h[2] = (__bf16)rb[i].z; cb.h[3] = (__bf16)rb[i].w;
      *(uint2*)&As[buf][r * LDK + c4 * 4] = ca.u;
      *(uint2*)&Bs[buf][r * LDK + c4 * 4] = cb.u;
    }
  };

  gload(0);
  lstore(0);
  __syncthreads();

  const int nk = K / BK;
  for (int kt = 0; kt < nk; ++kt) {
    const int cur = kt & 1;
    if (kt + 1 < nk) gload(kt + 1);               // prefetch next k-slice (HBM)

    v16bf bf0 = ld_frag_B(&Bs[cur][(wn * 32 + 0 * 16 + l16) * LDK], half);
    v16bf bf1 = ld_frag_B(&Bs[cur][(wn * 32 + 1 * 16 + l16) * LDK], half);
#pragma unroll
    for (int fm = 0; fm < 4; ++fm) {
      v16bf a = ld_frag_A(&As[cur][(wm * 64 + fm * 16 + l16) * LDK], half);
      acc[fm][0] = __builtin_amdgcn_wmma_f32_16x16x32_bf16(
          false, a, false, bf0, (short)0, acc[fm][0], false, false);
      acc[fm][1] = __builtin_amdgcn_wmma_f32_16x16x32_bf16(
          false, a, false, bf1, (short)0, acc[fm][1], false, false);
    }

    if (kt + 1 < nk) lstore((kt + 1) & 1);        // prior readers of buf^1 synced last iter
    __syncthreads();
  }

  // ---- fused epilogue: C layout = VGPR r -> M = r + 8*half, N = l16 ----
  const int m0 = by * BM + wm * 64 + half * 8;
  const int n0 = bx * BN + wn * 32;
#pragma unroll
  for (int fm = 0; fm < 4; ++fm) {
#pragma unroll
    for (int fn = 0; fn < 2; ++fn) {
      const int ncol = n0 + fn * 16 + l16;
      if (MODE == 0) {
        const int o = ncol >> 3;                  // NC == 8
        const int c = ncol & 7;
        const float bv = bias ? bias[ncol] : 0.0f;
#pragma unroll
        for (int r = 0; r < 8; ++r) {
          const int m = m0 + fm * 16 + r;
          const size_t oi = (size_t)m * O + o;
          if ((int)ctx[oi] == c) out[oi] = acc[fm][fn][r] + bv;  // 1-of-8 select
        }
      } else {
        const float thr = Tg[ncol];
#pragma unroll
        for (int r = 0; r < 8; ++r) {
          const int m = m0 + fm * 16 + r;
          bits[(size_t)m * N + ncol] = acc[fm][fn][r] > thr ? (uint8_t)1 : (uint8_t)0;
        }
      }
    }
  }
}

// ctx[b][o] = bits[b][3o] | bits[b][3o+1]<<1 | bits[b][3o+2]<<2   (pow2 = [1,2,4])
__global__ __launch_bounds__(256)
void pack_ctx(const uint8_t* __restrict__ bits, uint8_t* __restrict__ ctx,
              int total, int O)
{
  const int i = blockIdx.x * 256 + threadIdx.x;
  if (i >= total) return;
  const int b = i / O, o = i % O;
  const uint8_t* p = bits + (size_t)b * 3 * O + 3 * o;
  ctx[i] = (uint8_t)(p[0] | (p[1] << 1) | (p[2] << 2));
}

extern "C" void kernel_launch(void* const* d_in, const int* in_sizes, int n_in,
                              void* d_out, int out_size, void* d_ws, size_t ws_size,
                              hipStream_t stream)
{
  (void)in_sizes; (void)n_in; (void)out_size; (void)ws_size;

  const float* x  = (const float*)d_in[0];
  const float* W0 = (const float*)d_in[1];
  const float* W1 = (const float*)d_in[2];
  const float* W2 = (const float*)d_in[3];
  const float* b1 = (const float*)d_in[4];
  const float* b2 = (const float*)d_in[5];
  const float* H0 = (const float*)d_in[6];
  const float* T0 = (const float*)d_in[7];
  const float* H1 = (const float*)d_in[8];
  const float* T1 = (const float*)d_in[9];
  const float* H2 = (const float*)d_in[10];
  const float* T2 = (const float*)d_in[11];

  constexpr int B = 4096, DIN = 1024, DLAT = 2048, DOUT = 1024;

  // workspace carve (~113 MB): h1, h2, ctx0/1/2, shared bit scratch
  char* p = (char*)d_ws;
  float*   h1   = (float*)p;   p += (size_t)B * DLAT * sizeof(float);
  float*   h2   = (float*)p;   p += (size_t)B * DLAT * sizeof(float);
  uint8_t* ctx0 = (uint8_t*)p; p += (size_t)B * DLAT;
  uint8_t* ctx1 = (uint8_t*)p; p += (size_t)B * DLAT;
  uint8_t* ctx2 = (uint8_t*)p; p += (size_t)B * DOUT;
  uint8_t* bits = (uint8_t*)p; p += (size_t)B * 3 * DLAT;

  const dim3 blk(256);
  const dim3 gLat(3 * DLAT / BN, B / BM);   // 48 x 32
  const dim3 gOut(3 * DOUT / BN, B / BM);   // 24 x 32
  const dim3 gW01(8 * DLAT / BN, B / BM);   // 128 x 32
  const dim3 gW2 (8 * DOUT / BN, B / BM);   // 64 x 32
  const dim3 gP1((B * DLAT + 255) / 256);
  const dim3 gP2((B * DOUT + 255) / 256);

  // contexts (z == original input x for every layer)
  gln_gemm<1><<<gLat, blk, 0, stream>>>(x, H0, nullptr, nullptr, T0, bits, nullptr,
                                        DIN, 3 * DLAT, DLAT);
  pack_ctx<<<gP1, blk, 0, stream>>>(bits, ctx0, B * DLAT, DLAT);

  gln_gemm<1><<<gLat, blk, 0, stream>>>(x, H1, nullptr, nullptr, T1, bits, nullptr,
                                        DIN, 3 * DLAT, DLAT);
  pack_ctx<<<gP1, blk, 0, stream>>>(bits, ctx1, B * DLAT, DLAT);

  gln_gemm<1><<<gOut, blk, 0, stream>>>(x, H2, nullptr, nullptr, T2, bits, nullptr,
                                        DIN, 3 * DOUT, DOUT);
  pack_ctx<<<gP2, blk, 0, stream>>>(bits, ctx2, B * DOUT, DOUT);

  // layers: dense WMMA over all 8 experts, select fused in epilogue
  gln_gemm<0><<<gW01, blk, 0, stream>>>(x,  W0, nullptr, ctx0, nullptr, nullptr, h1,
                                        DIN,  8 * DLAT, DLAT);
  gln_gemm<0><<<gW01, blk, 0, stream>>>(h1, W1, b1,      ctx1, nullptr, nullptr, h2,
                                        DLAT, 8 * DLAT, DLAT);
  gln_gemm<0><<<gW2,  blk, 0, stream>>>(h2, W2, b2,      ctx2, nullptr, nullptr,
                                        (float*)d_out, DLAT, 8 * DOUT, DOUT);
}
